// AttentionBlock_31241592111542
// MI455X (gfx1250) — matrix operand order repaired and checked by reading, not compiled, not run on hardware
//
#include <hip/hip_runtime.h>
#include <hip/hip_bf16.h>

// ---------------------------------------------------------------------------
// AttentionBlock for MI455X (gfx1250).
// All four GEMMs use v_wmma_f32_16x16x32_bf16 with a 64x64-per-wave tile:
// 16 WMMAs per K-step reusing 4 A-frags x 4 B-frags from registers
// (~1.0 global_load_b128 per WMMA instead of 2.5 with a 16-wide tile).
// Compute-bound workload (AI > 1000 FLOP/B); L2 (192 MB) holds all operands.
// ---------------------------------------------------------------------------

typedef __attribute__((ext_vector_type(16))) __bf16 v16bf;
typedef __attribute__((ext_vector_type(8)))  __bf16 v8bf;
typedef __attribute__((ext_vector_type(8)))  float  v8f;

#define WMMA_BF16(a, b, c) \
  __builtin_amdgcn_wmma_f32_16x16x32_bf16(false, (a), false, (b), (short)0, (c), false, false)

static constexpr int BB = 8;      // batch
static constexpr int TT = 2048;   // sequence
static constexpr int DD = 1024;   // model dim (= K = V)

#define NEG_INF (-__builtin_huge_valf())

// ---- A-fragment (16x32 bf16, row-major source, row stride ld elements) ----
// Contiguous K order (same as B frag): lanes 0-15 row M=lane, K 0-15; lanes 16-31 K 16-31.
__device__ __forceinline__ v16bf load_a_frag(const __bf16* __restrict__ base,
                                             size_t ld, int lane) {
  const int row = lane & 15, half = lane >> 4;
  return *(const v16bf*)(base + (size_t)row * ld + half * 16);
}

// ---- B-fragment (32x16 bf16) from B^T stored row-major [N, Kdim] ----------
// lanes 0-15: col N=lane, K 0-15; lanes 16-31: col N=lane-16, K 16-31.
__device__ __forceinline__ v16bf load_bt_frag(const __bf16* __restrict__ bt,
                                              size_t ld, int lane) {
  const int col = lane & 15, half = lane >> 4;
  return *(const v16bf*)(bt + (size_t)col * ld + half * 16);
}

// ---- one K=32 step of the 64x64 wave tile: 16 WMMAs, 8 frag loads ---------
__device__ __forceinline__ void mma_step_64x64(const __bf16* __restrict__ A0,
                                               size_t lda,
                                               const __bf16* __restrict__ B0,
                                               size_t ldb, int lane,
                                               v8f acc[4][4]) {
  v16bf af[4];
#pragma unroll
  for (int i = 0; i < 4; ++i)
    af[i] = load_a_frag(A0 + (size_t)(i * 16) * lda, lda, lane);
#pragma unroll
  for (int j = 0; j < 4; ++j) {
    const v16bf bf_ = load_bt_frag(B0 + (size_t)(j * 16) * ldb, ldb, lane);
#pragma unroll
    for (int i = 0; i < 4; ++i)
      acc[i][j] = WMMA_BF16(af[i], bf_, acc[i][j]);
  }
}

// ===========================================================================
// Kernel 1: f32 -> bf16 conversion (weights and X)
// ===========================================================================
__global__ void __launch_bounds__(256)
cvt_bf16_kernel(const float* __restrict__ src, __bf16* __restrict__ dst,
                size_t n) {
  for (size_t i = (size_t)blockIdx.x * blockDim.x + threadIdx.x; i < n;
       i += (size_t)gridDim.x * blockDim.x)
    dst[i] = (__bf16)src[i];
}

// ===========================================================================
// Kernel 2: QKV projections.  grid = (16384/64, 1024/512, 3), block = 256.
// Each wave: 64x64 output tile over K=1024.  V stored transposed Vt[b,v,t].
// ===========================================================================
__global__ void __launch_bounds__(256)
qkv_gemm_kernel(const __bf16* __restrict__ Xb,
                const __bf16* __restrict__ Wq, const __bf16* __restrict__ Wk,
                const __bf16* __restrict__ Wv,
                const float* __restrict__ bq, const float* __restrict__ bk,
                const float* __restrict__ bv,
                __bf16* __restrict__ Qo, __bf16* __restrict__ Ko,
                __bf16* __restrict__ Vt) {
  const int tid = threadIdx.x, lane = tid & 31, wave = tid >> 5;
  const int m0 = blockIdx.x * 64;                 // flattened (b*T + t) rows
  const int n0 = blockIdx.y * 512 + wave * 64;    // output feature cols
  const int z  = blockIdx.z;                      // 0=Q 1=K 2=V
  const __bf16* W    = (z == 0) ? Wq : (z == 1) ? Wk : Wv;
  const float*  bias = (z == 0) ? bq : (z == 1) ? bk : bv;

  v8f acc[4][4] = {};
  const __bf16* A = Xb + (size_t)m0 * DD;
  const __bf16* B = W + (size_t)n0 * DD;
  for (int d0 = 0; d0 < DD; d0 += 32)
    mma_step_64x64(A + d0, DD, B + d0, DD, lane, acc);

  const int ncol = lane & 15, mb = (lane >> 4) * 8;
  for (int j = 0; j < 4; ++j) {
    const int n = n0 + j * 16 + ncol;
    const float bsc = bias[n];
    for (int i = 0; i < 4; ++i) {
#pragma unroll
      for (int r = 0; r < 8; ++r) {
        const int row = m0 + i * 16 + mb + r;
        const float val = acc[i][j][r] + bsc;
        if (z == 0) {
          Qo[(size_t)row * DD + n] = (__bf16)val;
        } else if (z == 1) {
          Ko[(size_t)row * DD + n] = (__bf16)val;
        } else {
          const int bidx = row >> 11, t = row & (TT - 1);
          Vt[((size_t)bidx * DD + n) * TT + t] = (__bf16)val;  // transposed
        }
      }
    }
  }
}

// ===========================================================================
// Kernel 3: logits S[b,t,s] = Q[b,t,:]·K[b,s,:] with causal mask.
// grid = (2048/64, 2048/512, 8).  Waves fully above the diagonal skip work.
// ===========================================================================
__global__ void __launch_bounds__(256)
logits_kernel(const __bf16* __restrict__ Q, const __bf16* __restrict__ K,
              float* __restrict__ S) {
  const int tid = threadIdx.x, lane = tid & 31, wave = tid >> 5;
  const int b  = blockIdx.z;
  const int t0 = blockIdx.x * 64;
  const int s0 = blockIdx.y * 512 + wave * 64;
  const int ncol = lane & 15, mb = (lane >> 4) * 8;
  float* Sb = S + (size_t)b * TT * TT;

  if (s0 > t0 + 63) {  // entire 64x64 tile strictly above the diagonal
    for (int i = 0; i < 4; ++i)
      for (int j = 0; j < 4; ++j)
#pragma unroll
        for (int r = 0; r < 8; ++r)
          Sb[(size_t)(t0 + i * 16 + mb + r) * TT + (s0 + j * 16 + ncol)] =
              NEG_INF;
    return;
  }

  v8f acc[4][4] = {};
  const __bf16* A = Q + ((size_t)b * TT + t0) * DD;
  const __bf16* B = K + ((size_t)b * TT + s0) * DD;
  for (int d0 = 0; d0 < DD; d0 += 32)
    mma_step_64x64(A + d0, DD, B + d0, DD, lane, acc);

  for (int i = 0; i < 4; ++i) {
    for (int j = 0; j < 4; ++j) {
#pragma unroll
      for (int r = 0; r < 8; ++r) {
        const int t = t0 + i * 16 + mb + r, s = s0 + j * 16 + ncol;
        Sb[(size_t)t * TT + s] = (s > t) ? NEG_INF : acc[i][j][r];
      }
    }
  }
}

// ===========================================================================
// Kernel 4: softmax over axis=1 (over t, per key column s), then *1/32.
// One thread per (b, s) column; online max/sum; writes bf16 probs.
// ===========================================================================
__global__ void __launch_bounds__(256)
softmax_col_kernel(const float* __restrict__ S, __bf16* __restrict__ P) {
  const int idx = blockIdx.x * 256 + threadIdx.x;  // 8*2048 columns
  const int b = idx >> 11, s = idx & (TT - 1);
  const float* cs = S + (size_t)b * TT * TT + s;
  float m = NEG_INF, sum = 0.0f;
  for (int t = s; t < TT; ++t) {  // rows t < s are masked
    const float x = cs[(size_t)t * TT];
    if (x > m) { sum *= __expf(m - x); m = x; }
    sum += __expf(x - m);
  }
  const float inv = 1.0f / (sum * 32.0f);  // post-softmax 1/sqrt(1024)
  __bf16* cp = P + (size_t)b * TT * TT + s;
  for (int t = 0; t < s; ++t) cp[(size_t)t * TT] = (__bf16)0.0f;
  for (int t = s; t < TT; ++t)
    cp[(size_t)t * TT] = (__bf16)(__expf(cs[(size_t)t * TT] - m) * inv);
}

// ===========================================================================
// Kernel 5: read = P · V  (+ residual X).  grid = (2048/64, 1024/512, 8).
// Reduction over s truncated at t0+64 (probs are zero for s > t).
// ===========================================================================
__global__ void __launch_bounds__(256)
read_gemm_kernel(const __bf16* __restrict__ P, const __bf16* __restrict__ Vt,
                 const float* __restrict__ X, float* __restrict__ Out) {
  const int tid = threadIdx.x, lane = tid & 31, wave = tid >> 5;
  const int b  = blockIdx.z;
  const int t0 = blockIdx.x * 64;
  const int v0 = blockIdx.y * 512 + wave * 64;
  const int ncol = lane & 15, mb = (lane >> 4) * 8;

  const __bf16* A = P + ((size_t)b * TT + t0) * TT;    // strip [64, 2048]
  const __bf16* B = Vt + ((size_t)b * DD + v0) * TT;   // B^T rows [v, s]

  const int dend = t0 + 64;  // probs vanish for s > t0+63
  v8f acc[4][4] = {};
  for (int d0 = 0; d0 < dend; d0 += 32)
    mma_step_64x64(A + d0, TT, B + d0, TT, lane, acc);

  for (int i = 0; i < 4; ++i) {
    for (int j = 0; j < 4; ++j) {
#pragma unroll
      for (int r = 0; r < 8; ++r) {
        const int t = t0 + i * 16 + mb + r, v = v0 + j * 16 + ncol;
        const size_t o = ((size_t)b * TT + t) * DD + v;
        Out[o] = X[o] + acc[i][j][r];
      }
    }
  }
}

// ===========================================================================
extern "C" void kernel_launch(void* const* d_in, const int* in_sizes, int n_in,
                              void* d_out, int out_size, void* d_ws, size_t ws_size,
                              hipStream_t stream) {
  const float* X  = (const float*)d_in[0];
  const float* Wq = (const float*)d_in[1];
  const float* bq = (const float*)d_in[2];
  const float* Wk = (const float*)d_in[3];
  const float* bk = (const float*)d_in[4];
  const float* Wv = (const float*)d_in[5];
  const float* bv = (const float*)d_in[6];
  float* Out = (float*)d_out;

  char* ws = (char*)d_ws;
  size_t off = 0;
  auto alloc = [&](size_t bytes) -> char* {
    char* p = ws + off;
    off += (bytes + 255) & ~(size_t)255;
    return p;
  };

  const size_t wn = (size_t)DD * DD;        // weight elements
  const size_t xn = (size_t)BB * TT * DD;   // activation elements
  const size_t sn = (size_t)BB * TT * TT;   // attention-matrix elements

  __bf16* Wqb = (__bf16*)alloc(wn * sizeof(__bf16));
  __bf16* Wkb = (__bf16*)alloc(wn * sizeof(__bf16));
  __bf16* Wvb = (__bf16*)alloc(wn * sizeof(__bf16));
  __bf16* Xb  = (__bf16*)alloc(xn * sizeof(__bf16));
  __bf16* Qb  = (__bf16*)alloc(xn * sizeof(__bf16));
  __bf16* Kb  = (__bf16*)alloc(xn * sizeof(__bf16));
  __bf16* Vtb = (__bf16*)alloc(xn * sizeof(__bf16));
  float*  Sb  = (float*)alloc(sn * sizeof(float));
  __bf16* Pb  = (__bf16*)alloc(sn * sizeof(__bf16));
  (void)ws_size; (void)in_sizes; (void)n_in; (void)out_size;

  cvt_bf16_kernel<<<dim3(256), dim3(256), 0, stream>>>(Wq, Wqb, wn);
  cvt_bf16_kernel<<<dim3(256), dim3(256), 0, stream>>>(Wk, Wkb, wn);
  cvt_bf16_kernel<<<dim3(256), dim3(256), 0, stream>>>(Wv, Wvb, wn);
  cvt_bf16_kernel<<<dim3(1024), dim3(256), 0, stream>>>(X, Xb, xn);

  qkv_gemm_kernel<<<dim3(256, 2, 3), dim3(256), 0, stream>>>(
      Xb, Wqb, Wkb, Wvb, bq, bk, bv, Qb, Kb, Vtb);

  logits_kernel<<<dim3(TT / 64, 4, BB), dim3(256), 0, stream>>>(Qb, Kb, Sb);

  softmax_col_kernel<<<dim3((BB * TT) / 256), dim3(256), 0, stream>>>(Sb, Pb);

  read_gemm_kernel<<<dim3(TT / 64, 2, BB), dim3(256), 0, stream>>>(Pb, Vtb, X, Out);
}
